// RetinaNet_50775103373501
// MI455X (gfx1250) — matrix operand order repaired and checked by reading, not compiled, not run on hardware
//
#include <hip/hip_runtime.h>

typedef __bf16 bf16_t;
typedef bf16_t v16bf __attribute__((ext_vector_type(16)));
typedef bf16_t v8bf  __attribute__((ext_vector_type(8)));
typedef float  v8f   __attribute__((ext_vector_type(8)));

#define BLK_PIX 128
#define BLK_OCH 64
#define CIN     256

// ---------------------------------------------------------------------------
// CDNA5 async global->LDS copy (16B per lane) + asynccnt wait, via inline asm
// (bypasses the toolchain-dependent builtin signatures; see CDNA5 ISA).
// VDST = per-lane LDS byte address, VADDR = per-lane 64-bit global address.
// ---------------------------------------------------------------------------
__device__ __forceinline__ void async_copy_b128(void* lds_dst, const void* gsrc) {
    asm volatile("global_load_async_to_lds_b128 %0, %1, off"
                 :: "v"((unsigned)(uintptr_t)lds_dst),
                    "v"((unsigned long long)(uintptr_t)gsrc)
                 : "memory");
}
__device__ __forceinline__ void wait_async() {
    asm volatile("s_wait_asynccnt 0x0" ::: "memory");
}

// ---------------------------------------------------------------------------
// Pack conv weights: [Orows][C][3][3] fp32 -> [Orows][9][C] bf16
// ---------------------------------------------------------------------------
__global__ void pack_weights_bf16(const float* __restrict__ w,
                                  bf16_t* __restrict__ out,
                                  long long total, int C) {
    long long idx = (long long)blockIdx.x * 256 + threadIdx.x;
    if (idx >= total) return;
    int c = (int)(idx % C);
    long long t = idx / C;
    int tap = (int)(t % 9);
    long long o = t / 9;
    out[idx] = (bf16_t)w[(o * C + c) * 9 + tap];
}

// ---------------------------------------------------------------------------
// NCHW fp32 -> NHWC bf16 (coalesced writes)
// ---------------------------------------------------------------------------
__global__ void nchw_to_nhwc_bf16(const float* __restrict__ in,
                                  bf16_t* __restrict__ out,
                                  int C, int HW, int N) {
    long long idx = (long long)blockIdx.x * 256 + threadIdx.x;
    long long total = (long long)N * HW * C;
    if (idx >= total) return;
    int c = (int)(idx % C);
    long long t = idx / C;
    int p = (int)(t % HW);
    int n = (int)(t / HW);
    out[idx] = (bf16_t)in[((long long)(n * C + c)) * HW + p];
}

// ---------------------------------------------------------------------------
// Implicit-GEMM 3x3 SAME conv via v_wmma_f32_16x16x32_bf16.
//   A = weights [O x K], B = activations [K x HW], K = 9*CIN.
// Double-buffered LDS staged with async global->LDS DMA.
// WD (= H = W) is a compile-time constant per FPN level.
// ---------------------------------------------------------------------------
template<int WD>
__global__ __launch_bounds__(256)
void conv3x3_wmma_bf16(const bf16_t* __restrict__ x,   // [N][WD][WD][CIN] bf16
                       const bf16_t* __restrict__ wp,  // [O][9][CIN] bf16
                       const float*  __restrict__ bias,// [O]
                       bf16_t* __restrict__ y_relu,    // NHWC bf16 or null
                       float*  __restrict__ y_fin,     // final fp32 or null
                       int O, int Kk, int Anch,
                       long long fin_stride_n, long long fin_off) {
    constexpr int HW = WD * WD;
    constexpr int NCHUNK = 9 * (CIN / 32);   // 72 k-chunks of 32

    __shared__ bf16_t Alds[2][BLK_OCH * 32]; // weights  [o_local][c_local]
    __shared__ bf16_t Blds[2][BLK_PIX * 32]; // activ.   [pix_local][c_local]

    const int t    = threadIdx.x;
    const int lane = t & 31;
    const int wave = t >> 5;
    const int wm   = wave & 1;   // 2 waves over 64 out-channels
    const int wn   = wave >> 1;  // 4 waves over 128 pixels
    const int nb   = blockIdx.z;
    const int p0   = blockIdx.x * BLK_PIX;
    const int o0   = blockIdx.y * BLK_OCH;

    v8f acc[2][2];
    acc[0][0] = (v8f){}; acc[0][1] = (v8f){};
    acc[1][0] = (v8f){}; acc[1][1] = (v8f){};

    // stage one 32-channel K-chunk (tap = kc/8, c0 = (kc%8)*32) into buffer buf
    auto stage = [&](int buf, int kc) {
        const int tap = kc >> 3;
        const int c0  = (kc & 7) * 32;
        const int dr  = tap / 3 - 1;
        const int dc  = tap % 3 - 1;
        {   // weights: 64 rows x 32 ch = 256 x 16B units, one per thread
            int ol = t >> 2, part = t & 3;
            int o = o0 + ol;
            bf16_t* dst = &Alds[buf][ol * 32 + part * 8];
            if (o < O) {
                async_copy_b128(dst, wp + ((long long)o * 9 + tap) * CIN + c0 + part * 8);
            } else {
                uint4 z = {0u, 0u, 0u, 0u};
                *(uint4*)dst = z;
            }
        }
        // shifted activations: 128 px x 32 ch = 512 x 16B units
        for (int u = t; u < BLK_PIX * 4; u += 256) {
            int pl = u >> 2, part = u & 3;
            int p = p0 + pl;
            bf16_t* dst = &Blds[buf][pl * 32 + part * 8];
            const bf16_t* src = nullptr;
            if (p < HW) {
                int h = p / WD, ww = p - h * WD;   // WD compile-time: no div
                int hh = h + dr, wc = ww + dc;
                if ((unsigned)hh < (unsigned)WD && (unsigned)wc < (unsigned)WD)
                    src = x + ((long long)nb * HW + hh * WD + wc) * CIN + c0 + part * 8;
            }
            if (src) {
                async_copy_b128(dst, src);
            } else {
                uint4 z = {0u, 0u, 0u, 0u};
                *(uint4*)dst = z;
            }
        }
    };

    const int mrow0 = wm * 32 + (lane & 15);
    const int prow0 = wn * 32 + (lane & 15);
    const int ka    = (lane < 16) ? 0 : 16;   // A: lanes 0-15 K=0..15
    const int kb    = (lane < 16) ? 0 : 16;   // B: lanes 0-15 K=0..15

    stage(0, 0);
    wait_async();
    __syncthreads();

    for (int kc = 0; kc < NCHUNK; ++kc) {
        const int cur = kc & 1;
        if (kc + 1 < NCHUNK)
            stage(cur ^ 1, kc + 1);           // async prefetch next chunk

        // fragments in contiguous k order (k = 16*(lane/16) + s), same for A and B
        v16bf afrag[2], bfrag[2];
        for (int mi = 0; mi < 2; ++mi) {
            const bf16_t* base = &Alds[cur][(mrow0 + mi * 16) * 32];
            v8bf lo = *(const v8bf*)(base + ka);
            v8bf hi = *(const v8bf*)(base + ka + 8);
            for (int i = 0; i < 8; ++i) {
                afrag[mi][i]     = lo[i];
                afrag[mi][i + 8] = hi[i];
            }
        }
        for (int ni = 0; ni < 2; ++ni)
            bfrag[ni] = *(const v16bf*)(&Blds[cur][(prow0 + ni * 16) * 32 + kb]);

        for (int mi = 0; mi < 2; ++mi)
            for (int ni = 0; ni < 2; ++ni)
                acc[mi][ni] = __builtin_amdgcn_wmma_f32_16x16x32_bf16(
                    false, afrag[mi], false, bfrag[ni],
                    (short)0, acc[mi][ni], false, false);

        wait_async();                          // next-chunk copies landed
        __syncthreads();                       // all waves done reading cur
    }

    // epilogue: lane = fixed pixel, 8 consecutive channels per accumulator reg
    const int nloc = lane & 15;
    const int hi8  = (lane >> 4) * 8;
    for (int ni = 0; ni < 2; ++ni) {
        int p = p0 + wn * 32 + ni * 16 + nloc;
        if (p >= HW) continue;
        for (int mi = 0; mi < 2; ++mi) {
            int chb = o0 + wm * 32 + mi * 16 + hi8;
            if (y_relu) {
                alignas(16) bf16_t vals[8];
                for (int r = 0; r < 8; ++r) {
                    float v = acc[mi][ni][r] + bias[chb + r];
                    v = v > 0.f ? v : 0.f;
                    vals[r] = (bf16_t)v;
                }
                *(uint4*)(y_relu + ((long long)nb * HW + p) * O + chb) =
                    *(const uint4*)vals;
            } else {
                for (int r = 0; r < 8; ++r) {
                    int o = chb + r;
                    if (o < O) {
                        float v = acc[mi][ni][r] + bias[o];
                        int a = o / Kk, k = o - a * Kk;
                        y_fin[(long long)nb * fin_stride_n + fin_off
                              + (long long)p * Anch * Kk + (long long)a * Kk + k] = v;
                    }
                }
            }
        }
    }
}

// ---------------------------------------------------------------------------
// Host: per-level launcher (WD compile-time)
// ---------------------------------------------------------------------------
template<int WD>
static void run_level(const float* feat,
                      bf16_t* in_bf, bf16_t* ping, bf16_t* pong,
                      bf16_t* cls_tw, const float* cls_cb,
                      bf16_t* cls_ow, const float* cls_ob,
                      bf16_t* reg_tw, const float* reg_cb,
                      bf16_t* reg_ow, const float* reg_ob,
                      float* cls_base, float* reg_base,
                      long long off_l, long long TOT, hipStream_t stream) {
    constexpr int HW = WD * WD;
    const int N = 4, A = 9;
    {
        long long total = (long long)N * HW * CIN;
        dim3 g((unsigned)((total + 255) / 256));
        nchw_to_nhwc_bf16<<<g, 256, 0, stream>>>(feat, in_bf, CIN, HW, N);
    }
    for (int br = 0; br < 2; ++br) {
        bf16_t* tw        = br ? reg_tw : cls_tw;
        const float* tb   = br ? reg_cb : cls_cb;
        bf16_t* ow        = br ? reg_ow : cls_ow;
        const float* ob   = br ? reg_ob : cls_ob;
        const int Ofin    = br ? 36 : 720;
        const int Kk      = br ? 4 : 80;
        float* fbase      = br ? reg_base : cls_base;
        const long long strn = TOT * (long long)Kk;

        bf16_t* cur = in_bf;
        bf16_t* nxt = ping;
        dim3 gt((unsigned)((HW + BLK_PIX - 1) / BLK_PIX),
                (unsigned)((CIN + BLK_OCH - 1) / BLK_OCH), (unsigned)N);
        for (int i = 0; i < 4; ++i) {
            conv3x3_wmma_bf16<WD><<<gt, 256, 0, stream>>>(
                cur, tw + (size_t)i * CIN * 9 * CIN, tb + i * CIN,
                nxt, nullptr, CIN, 1, 1, 0, 0);
            cur = nxt;
            nxt = (cur == ping) ? pong : ping;
        }
        dim3 gf((unsigned)((HW + BLK_PIX - 1) / BLK_PIX),
                (unsigned)((Ofin + BLK_OCH - 1) / BLK_OCH), (unsigned)N);
        conv3x3_wmma_bf16<WD><<<gf, 256, 0, stream>>>(
            cur, ow, ob, nullptr, fbase,
            Ofin, Kk, A, strn, off_l * Kk);
    }
}

extern "C" void kernel_launch(void* const* d_in, const int* in_sizes, int n_in,
                              void* d_out, int out_size, void* d_ws, size_t ws_size,
                              hipStream_t stream) {
    (void)in_sizes; (void)n_in; (void)out_size; (void)ws_size;
    const float* feat[5];
    for (int i = 0; i < 5; ++i) feat[i] = (const float*)d_in[i];
    const float* cls_conv_w = (const float*)d_in[5];
    const float* cls_conv_b = (const float*)d_in[6];
    const float* cls_out_w  = (const float*)d_in[7];
    const float* cls_out_b  = (const float*)d_in[8];
    const float* reg_conv_w = (const float*)d_in[9];
    const float* reg_conv_b = (const float*)d_in[10];
    const float* reg_out_w  = (const float*)d_in[11];
    const float* reg_out_b  = (const float*)d_in[12];

    const int N = 4;
    const long long off_l[5] = {0, 90000, 112500, 118125, 119646};
    const long long TOT = 120087;   // sum(H*W*A)

    char* ws = (char*)d_ws;
    size_t off = 0;
    auto carve = [&](size_t bytes) -> void* {
        void* p = ws + off;
        off = (off + bytes + 255) & ~(size_t)255;
        return p;
    };
    const size_t tower_w_elems = 4LL * CIN * 9 * CIN;
    bf16_t* cls_tw = (bf16_t*)carve(tower_w_elems * 2);
    bf16_t* reg_tw = (bf16_t*)carve(tower_w_elems * 2);
    const size_t cls_ow_elems = 720LL * 9 * CIN;
    const size_t reg_ow_elems = 36LL * 9 * CIN;
    bf16_t* cls_ow = (bf16_t*)carve(cls_ow_elems * 2);
    bf16_t* reg_ow = (bf16_t*)carve(reg_ow_elems * 2);
    const size_t act_elems = (size_t)N * 100 * 100 * CIN;
    bf16_t* in_bf = (bf16_t*)carve(act_elems * 2);
    bf16_t* ping  = (bf16_t*)carve(act_elems * 2);
    bf16_t* pong  = (bf16_t*)carve(act_elems * 2);

    auto launch_pack = [&](const float* src, bf16_t* dst, long long total) {
        dim3 g((unsigned)((total + 255) / 256));
        pack_weights_bf16<<<g, 256, 0, stream>>>(src, dst, total, CIN);
    };
    launch_pack(cls_conv_w, cls_tw, (long long)tower_w_elems);
    launch_pack(reg_conv_w, reg_tw, (long long)tower_w_elems);
    launch_pack(cls_out_w, cls_ow, (long long)cls_ow_elems);
    launch_pack(reg_out_w, reg_ow, (long long)reg_ow_elems);

    float* cls_base = (float*)d_out;
    float* reg_base = (float*)d_out + (long long)N * TOT * 80;

    for (int l = 0; l < 5; ++l) {
        switch (l) {
        case 0: run_level<100>(feat[l], in_bf, ping, pong, cls_tw, cls_conv_b,
                               cls_ow, cls_out_b, reg_tw, reg_conv_b, reg_ow,
                               reg_out_b, cls_base, reg_base, off_l[l], TOT, stream);
                break;
        case 1: run_level<50>(feat[l], in_bf, ping, pong, cls_tw, cls_conv_b,
                              cls_ow, cls_out_b, reg_tw, reg_conv_b, reg_ow,
                              reg_out_b, cls_base, reg_base, off_l[l], TOT, stream);
                break;
        case 2: run_level<25>(feat[l], in_bf, ping, pong, cls_tw, cls_conv_b,
                              cls_ow, cls_out_b, reg_tw, reg_conv_b, reg_ow,
                              reg_out_b, cls_base, reg_base, off_l[l], TOT, stream);
                break;
        case 3: run_level<13>(feat[l], in_bf, ping, pong, cls_tw, cls_conv_b,
                              cls_ow, cls_out_b, reg_tw, reg_conv_b, reg_ow,
                              reg_out_b, cls_base, reg_base, off_l[l], TOT, stream);
                break;
        case 4: run_level<7>(feat[l], in_bf, ping, pong, cls_tw, cls_conv_b,
                             cls_ow, cls_out_b, reg_tw, reg_conv_b, reg_ow,
                             reg_out_b, cls_base, reg_base, off_l[l], TOT, stream);
                break;
        }
    }
}
